// DynamicsPredictor_46712064311370
// MI455X (gfx1250) — compile-verified, matrix operand order checked by reading, not run on hardware
//
#include <hip/hip_runtime.h>
#include <hip/hip_bf16.h>

typedef __attribute__((ext_vector_type(16))) _Float16 v16h;
typedef __attribute__((ext_vector_type(8)))  _Float16 v8h;
typedef __attribute__((ext_vector_type(8)))  float    v8f;

#define B_ 128
#define T_ 128
#define H_ 512
#define S_ 512
#define E_ 512
#define A_ 32
#define TM_ 127

// d_out flat layout (floats), in reference return order
#define OUT_HID 0UL
#define OUT_PRI 8388608UL      // B*T*H
#define OUT_POS 16777216UL
#define OUT_PM  25165824UL
#define OUT_PLV 33488896UL     // OUT_PM + B*(T-1)*S
#define OUT_QM  41811968UL
#define OUT_QLV 50135040UL

// ---------------- WMMA fragment helpers ----------------
// A (16x32 f16): lanes 0-15 hold M=lane, K in [k0,k0+8) U [k0+16,k0+24);
// lanes 16-31 hold same M rows, K in [k0+8,k0+16) U [k0+24,k0+32).
// All GEMMs are X @ W^T with W stored [N][K] row-major, so B fragments
// load with the identical pattern (N in place of M).
__device__ __forceinline__ v16h ldfrag(const _Float16* __restrict__ base, int ld,
                                       int row, int k0, int sel) {
    const _Float16* p = base + (size_t)row * ld + k0 + (sel << 3);
    v8h lo = *(const v8h*)p;          // 16B aligned
    v8h hi = *(const v8h*)(p + 16);   // +32B
    v16h f;
#pragma unroll
    for (int i = 0; i < 8; ++i) { f[i] = lo[i]; f[i + 8] = hi[i]; }
    return f;
}

__device__ __forceinline__ v8f wmma_acc(v16h a, v16h b, v8f c) {
    return __builtin_amdgcn_wmma_f32_16x16x32_f16(false, a, false, b, (short)0, c,
                                                  false, false);
}

// split-K GEMM tile: K-segment 1 from X1 (W cols [0,K1)), segment 2 from X2
// (W cols [K1, K1+K2)).  X2 may be a strided view into a [B,T,*] array.
__device__ __forceinline__ v8f gemm_tile2(const _Float16* __restrict__ X1, int lda1, int K1,
                                          const _Float16* __restrict__ X2, int lda2, int K2,
                                          const _Float16* __restrict__ W, int ldb,
                                          int m0, int n0) {
    int lane = threadIdx.x & 31;
    int sel = lane >> 4, r = lane & 15;
    v8f acc = {};
    for (int k0 = 0; k0 < K1; k0 += 32) {
        v16h a = ldfrag(X1, lda1, m0 + r, k0, sel);
        v16h b = ldfrag(W, ldb, n0 + r, k0, sel);
        acc = wmma_acc(a, b, acc);
    }
    for (int k0 = 0; k0 < K2; k0 += 32) {
        v16h a = ldfrag(X2, lda2, m0 + r, k0, sel);
        v16h b = ldfrag(W, ldb, n0 + r, K1 + k0, sel);
        acc = wmma_acc(a, b, acc);
    }
    return acc;
}

__device__ __forceinline__ float sigm_(float x) { return 1.0f / (1.0f + __expf(-x)); }

// ---------------- elementwise kernels ----------------
__global__ void f32tof16_kernel(const float* __restrict__ s, _Float16* __restrict__ d, int n) {
    int i = blockIdx.x * blockDim.x + threadIdx.x;
    if (i < n) d[i] = (_Float16)s[i];
}

__global__ void init_kernel(const float* __restrict__ ph, const float* __restrict__ ps,
                            const unsigned char* __restrict__ dones,
                            float* __restrict__ hC, _Float16* __restrict__ h_h,
                            _Float16* __restrict__ Xsa_m, float* __restrict__ out) {
    int i = blockIdx.x * blockDim.x + threadIdx.x;
    if (i >= B_ * H_) return;
    int b = i >> 9, j = i & 511;
    float h = ph[i], s = ps[i];
    hC[i] = h;
    h_h[i] = (_Float16)h;
    float nd0 = dones[b * T_ + 0] ? 0.0f : 1.0f;   // step-0 mask
    Xsa_m[i] = (_Float16)(s * nd0);
    out[OUT_HID + ((size_t)b * T_) * H_ + j] = h;
    out[OUT_PRI + ((size_t)b * T_) * S_ + j] = s;
    out[OUT_POS + ((size_t)b * T_) * S_ + j] = s;
}

// ---------------- fused WMMA GEMM kernels ----------------
// bx<32 : sa = silu([s*nd, a_t] @ Wsa^T + b_sa)  (K = 512 + 32) -> f16
// bx>=32: gh = h_{t-1} @ Whh^T + b_hh            (N = 1536, K = 512) -> f32
__global__ void gemmAB_kernel(int t, const _Float16* __restrict__ Xsa_m,
                              const _Float16* __restrict__ act_h,
                              const _Float16* __restrict__ h_h,
                              const _Float16* __restrict__ hWsa, const _Float16* __restrict__ hWhh,
                              const float* __restrict__ b_sa, const float* __restrict__ b_hh,
                              _Float16* __restrict__ sa_h, float* __restrict__ gh) {
    int lane = threadIdx.x;
    int sel = lane >> 4;
    int m0 = blockIdx.y * 16;
    if (blockIdx.x < 32) {
        int n0 = blockIdx.x * 16;
        v8f acc = gemm_tile2(Xsa_m, S_, S_,
                             act_h + (size_t)t * A_, T_ * A_, A_,
                             hWsa, S_ + A_, m0, n0);
        int n = n0 + (lane & 15);
        float bb = b_sa[n];
#pragma unroll
        for (int v = 0; v < 8; ++v) {
            int m = m0 + sel * 8 + v;
            float x = acc[v] + bb;
            sa_h[m * H_ + n] = (_Float16)(x * sigm_(x));
        }
    } else {
        int n0 = (blockIdx.x - 32) * 16;
        v8f acc = gemm_tile2(h_h, H_, H_, nullptr, 0, 0, hWhh, H_, m0, n0);
        int n = n0 + (lane & 15);
        float bb = b_hh[n];
#pragma unroll
        for (int v = 0; v < 8; ++v) {
            int m = m0 + sel * 8 + v;
            gh[m * 1536 + n] = acc[v] + bb;
        }
    }
}

// GRU: one A-fragment feeds 3 WMMA chains (r,z,n gate columns j, j+512, j+1024),
// fused gating epilogue -> h_t (fp32 carry + single f16 buffer + output row t+1).
__global__ void gru_kernel(int t, const _Float16* __restrict__ sa_h,
                           const _Float16* __restrict__ hWih, const float* __restrict__ b_ih,
                           const float* __restrict__ gh, float* __restrict__ hC,
                           _Float16* __restrict__ h_h, float* __restrict__ out) {
    int lane = threadIdx.x;
    int sel = lane >> 4, r = lane & 15;
    int n0 = blockIdx.x * 16, m0 = blockIdx.y * 16;
    v8f ar = {}, az = {}, an = {};
    for (int k0 = 0; k0 < H_; k0 += 32) {
        v16h a  = ldfrag(sa_h, H_, m0 + r, k0, sel);
        v16h br = ldfrag(hWih,            H_, n0 + r, k0, sel);
        v16h bz = ldfrag(hWih + 512 * H_, H_, n0 + r, k0, sel);
        v16h bn = ldfrag(hWih + 1024 * H_, H_, n0 + r, k0, sel);
        ar = wmma_acc(a, br, ar);
        az = wmma_acc(a, bz, az);
        an = wmma_acc(a, bn, an);
    }
    int n = n0 + r;
    float bir = b_ih[n], biz = b_ih[n + 512], bin = b_ih[n + 1024];
#pragma unroll
    for (int v = 0; v < 8; ++v) {
        int m = m0 + sel * 8 + v;
        float hr = gh[m * 1536 + n];
        float hz = gh[m * 1536 + n + 512];
        float hn = gh[m * 1536 + n + 1024];
        float rg = sigm_(ar[v] + bir + hr);
        float zg = sigm_(az[v] + biz + hz);
        float ng = tanhf(an[v] + bin + rg * hn);
        float hnew = (1.0f - zg) * ng + zg * hC[m * H_ + n];
        hC[m * H_ + n] = hnew;
        h_h[m * H_ + n] = (_Float16)hnew;
        out[OUT_HID + ((size_t)m * T_ + (t + 1)) * H_ + n] = hnew;
    }
}

// bx<32 : ha = silu([h, a_t] @ Wha^T + b_ha)  K = 512+32
// bx>=32: ho = silu([h, o_t] @ Who^T + b_ho)  K = 512+512
__global__ void gemmCD_kernel(int t, const _Float16* __restrict__ h_h,
                              const _Float16* __restrict__ act_h,
                              const _Float16* __restrict__ obs_h,
                              const _Float16* __restrict__ hWha, const _Float16* __restrict__ hWho,
                              const float* __restrict__ b_ha, const float* __restrict__ b_ho,
                              _Float16* __restrict__ ha_h, _Float16* __restrict__ ho_h) {
    int lane = threadIdx.x;
    int sel = lane >> 4;
    int m0 = blockIdx.y * 16;
    const bool second = blockIdx.x >= 32;
    int n0 = (second ? blockIdx.x - 32 : blockIdx.x) * 16;
    v8f acc = second ? gemm_tile2(h_h, H_, H_,
                                  obs_h + (size_t)t * E_, T_ * E_, E_,
                                  hWho, H_ + E_, m0, n0)
                     : gemm_tile2(h_h, H_, H_,
                                  act_h + (size_t)t * A_, T_ * A_, A_,
                                  hWha, H_ + A_, m0, n0);
    const float* bias = second ? b_ho : b_ha;
    _Float16* dst = second ? ho_h : ha_h;
    int n = n0 + (lane & 15);
    float bb = bias[n];
#pragma unroll
    for (int v = 0; v < 8; ++v) {
        int m = m0 + sel * 8 + v;
        float x = acc[v] + bb;
        dst[m * H_ + n] = (_Float16)(x * sigm_(x));
    }
}

// Gaussian heads: one A-fragment feeds 2 WMMA chains (mu rows n, logvar rows n+512).
// bx<32 -> prior; bx>=32 -> posterior, which also emits the next step's masked
// f16 state f16(st * nd_{t+1}) straight into the sa-GEMM input buffer.
__global__ void head_kernel(int t, const _Float16* __restrict__ ha_h,
                            const _Float16* __restrict__ ho_h,
                            const _Float16* __restrict__ hWpri, const _Float16* __restrict__ hWpost,
                            const float* __restrict__ b_pri, const float* __restrict__ b_post,
                            const float* __restrict__ prior_noise,
                            const float* __restrict__ post_noise,
                            const unsigned char* __restrict__ dones,
                            _Float16* __restrict__ Xsa_m, float* __restrict__ out) {
    int lane = threadIdx.x;
    int sel = lane >> 4, r = lane & 15;
    const bool post = blockIdx.x >= 32;
    int n0 = (post ? blockIdx.x - 32 : blockIdx.x) * 16;
    int m0 = blockIdx.y * 16;
    const _Float16* X = post ? ho_h : ha_h;
    const _Float16* W = post ? hWpost : hWpri;
    const float* bias = post ? b_post : b_pri;
    const float* noise = post ? post_noise : prior_noise;
    v8f am = {}, al = {};
    for (int k0 = 0; k0 < H_; k0 += 32) {
        v16h a  = ldfrag(X, H_, m0 + r, k0, sel);
        v16h bm = ldfrag(W,            H_, n0 + r, k0, sel);
        v16h bl = ldfrag(W + 512 * H_, H_, n0 + r, k0, sel);
        am = wmma_acc(a, bm, am);
        al = wmma_acc(a, bl, al);
    }
    int n = n0 + r;
    float bmu = bias[n], blv = bias[n + 512];
    size_t OM  = post ? OUT_QM  : OUT_PM;
    size_t OLV = post ? OUT_QLV : OUT_PLV;
    size_t OST = post ? OUT_POS : OUT_PRI;
#pragma unroll
    for (int v = 0; v < 8; ++v) {
        int m = m0 + sel * 8 + v;
        float mu = am[v] + bmu;
        float lv = al[v] + blv;
        float nz = noise[((size_t)t * B_ + m) * S_ + n];
        // exp(softplus(lv)) == 1 + exp(lv)
        float st = mu + (1.0f + __expf(lv)) * nz;
        out[OM  + ((size_t)m * TM_ + t) * S_ + n] = mu;
        out[OLV + ((size_t)m * TM_ + t) * S_ + n] = lv;
        out[OST + ((size_t)m * T_ + (t + 1)) * S_ + n] = st;
        if (post) {
            float nd = dones[m * T_ + (t + 1)] ? 0.0f : 1.0f;
            Xsa_m[m * S_ + n] = (_Float16)(st * nd);
        }
    }
}

// ---------------- host launcher ----------------
extern "C" void kernel_launch(void* const* d_in, const int* in_sizes, int n_in,
                              void* d_out, int out_size, void* d_ws, size_t ws_size,
                              hipStream_t stream) {
    (void)in_sizes; (void)n_in; (void)out_size; (void)ws_size;
    const float* prev_hidden  = (const float*)d_in[0];
    const float* prev_state   = (const float*)d_in[1];
    const float* actions      = (const float*)d_in[2];
    const float* observations = (const float*)d_in[3];
    const unsigned char* dones = (const unsigned char*)d_in[4];
    const float* prior_noise  = (const float*)d_in[5];
    const float* post_noise   = (const float*)d_in[6];
    const float* W_sa = (const float*)d_in[7];   const float* b_sa = (const float*)d_in[8];
    const float* W_ih = (const float*)d_in[9];   const float* b_ih = (const float*)d_in[10];
    const float* W_hh = (const float*)d_in[11];  const float* b_hh = (const float*)d_in[12];
    const float* W_ha = (const float*)d_in[13];  const float* b_ha = (const float*)d_in[14];
    const float* W_pri = (const float*)d_in[15]; const float* b_pri = (const float*)d_in[16];
    const float* W_ho = (const float*)d_in[17];  const float* b_ho = (const float*)d_in[18];
    const float* W_post = (const float*)d_in[19]; const float* b_post = (const float*)d_in[20];
    float* out = (float*)d_out;

    char* ws = (char*)d_ws;
    size_t off = 0;
    auto take = [&](size_t bytes) -> char* {
        char* p = ws + off;
        off += (bytes + 255) & ~(size_t)255;
        return p;
    };
    _Float16* hWsa  = (_Float16*)take((size_t)512 * 544 * 2);
    _Float16* hWih  = (_Float16*)take((size_t)1536 * 512 * 2);
    _Float16* hWhh  = (_Float16*)take((size_t)1536 * 512 * 2);
    _Float16* hWha  = (_Float16*)take((size_t)512 * 544 * 2);
    _Float16* hWpri = (_Float16*)take((size_t)1024 * 512 * 2);
    _Float16* hWho  = (_Float16*)take((size_t)512 * 1024 * 2);   // W_ho is [512][1024]
    _Float16* hWpost= (_Float16*)take((size_t)1024 * 512 * 2);
    _Float16* act_h = (_Float16*)take((size_t)B_ * T_ * A_ * 2);
    _Float16* obs_h = (_Float16*)take((size_t)B_ * T_ * E_ * 2);
    _Float16* Xsa_m = (_Float16*)take((size_t)B_ * S_ * 2);
    _Float16* h_h   = (_Float16*)take((size_t)B_ * H_ * 2);
    _Float16* sa_h  = (_Float16*)take((size_t)B_ * H_ * 2);
    _Float16* ha_h  = (_Float16*)take((size_t)B_ * H_ * 2);
    _Float16* ho_h  = (_Float16*)take((size_t)B_ * H_ * 2);
    float* gh = (float*)take((size_t)B_ * 1536 * 4);
    float* hC = (float*)take((size_t)B_ * H_ * 4);

    // one-time f16 downconversion: weights + all timestep inputs (L2-resident)
    auto cv = [&](const float* s, _Float16* d, int n) {
        f32tof16_kernel<<<(n + 255) / 256, 256, 0, stream>>>(s, d, n);
    };
    cv(W_sa, hWsa, 512 * 544);
    cv(W_ih, hWih, 1536 * 512);
    cv(W_hh, hWhh, 1536 * 512);
    cv(W_ha, hWha, 512 * 544);
    cv(W_pri, hWpri, 1024 * 512);
    cv(W_ho, hWho, 512 * 1024);
    cv(W_post, hWpost, 1024 * 512);
    cv(actions, act_h, B_ * T_ * A_);
    cv(observations, obs_h, B_ * T_ * E_);

    init_kernel<<<(B_ * H_ + 255) / 256, 256, 0, stream>>>(
        prev_hidden, prev_state, dones, hC, h_h, Xsa_m, out);

    for (int t = 0; t < TM_; ++t) {
        gemmAB_kernel<<<dim3(128, 8), 32, 0, stream>>>(
            t, Xsa_m, act_h, h_h, hWsa, hWhh, b_sa, b_hh, sa_h, gh);
        gru_kernel<<<dim3(32, 8), 32, 0, stream>>>(
            t, sa_h, hWih, b_ih, gh, hC, h_h, out);
        gemmCD_kernel<<<dim3(64, 8), 32, 0, stream>>>(
            t, h_h, act_h, obs_h, hWha, hWho, b_ha, b_ho, ha_h, ho_h);
        head_kernel<<<dim3(64, 8), 32, 0, stream>>>(
            t, ha_h, ho_h, hWpri, hWpost, b_pri, b_post,
            prior_noise, post_noise, dones, Xsa_m, out);
    }
}
